// OptimizedCrossAttention_18004502905404
// MI455X (gfx1250) — compile-verified, matrix operand order checked by reading, not run on hardware
//
#include <hip/hip_runtime.h>

#define DEV __device__ __forceinline__

typedef __attribute__((ext_vector_type(16))) __bf16 v16bf;
typedef __attribute__((ext_vector_type(8)))  float  v8f;

union FragBF { v16bf v; unsigned u[8]; };
union Pack2  { __bf16 h[2]; unsigned u; };

DEV unsigned pack2f(float a, float b) {
  Pack2 p; p.h[0] = (__bf16)a; p.h[1] = (__bf16)b; return p.u;
}
// K-offset pattern for 16-bit A/B fragments (ISA 7.12.2):
// VGPR i<4 holds K = 2i,2i+1 (+8 for lanes 16..31); i>=4 holds K = 16+2(i-4),.. (+8)
DEV int kpat(int i, int hi) {
  return ((i < 4) ? (2 * i) : (16 + 2 * (i - 4))) + hi * 8;
}

DEV v8f wmma_bf16(const FragBF& a, const FragBF& b, v8f c) {
  return __builtin_amdgcn_wmma_f32_16x16x32_bf16(
      false, a.v, false, b.v, (short)0, c, false, false);
}

// gfx1250 async copy: 16B per lane, global -> LDS, tracked by ASYNCcnt.
DEV void async_ld_b128(unsigned lds_off, const void* gaddr) {
  asm volatile("global_load_async_to_lds_b128 %0, %1, off"
               :: "v"(lds_off), "v"(gaddr) : "memory");
}
DEV void wait_asynccnt0() {
  asm volatile("s_wait_asynccnt 0x0" ::: "memory");
}

static constexpr int Dm = 1024;

// ---------------------------------------------------------------------------
// GEMM: C[M,N] = A[M,K] @ W[N,K]^T + bias[N], K=N=1024.
// 128x128 tile / 256 threads (8 waves), BK=32, bf16 WMMA, fp32 accumulate.
// AT = float (convert to bf16 while staging) or __bf16 (straight copy).
// OUT_MODE: 0 = fp32 row-major, 1 = bf16 row-major,
//           2 = bf16 per-head transposed: Vt[(b*1024 + n)*2048 + t]
// ---------------------------------------------------------------------------
template <typename AT, int OUT_MODE>
__global__ __launch_bounds__(256) void gemm_bias_kernel(
    const AT* __restrict__ A, const float* __restrict__ W,
    const float* __restrict__ bias, void* __restrict__ Cout) {
  constexpr int LST = 40;  // LDS row stride in bf16 elements (bank-shift pad)
  __shared__ __bf16 sA[128 * LST];
  __shared__ __bf16 sW[128 * LST];

  const int tid   = threadIdx.x;
  const int lane  = tid & 31;
  const int wid   = tid >> 5;
  const int hi    = (lane >> 4) & 1;
  const int lrow  = lane & 15;
  const int waveM = (wid & 3) * 32;
  const int waveN = (wid >> 2) * 64;
  const int bM    = blockIdx.x * 128;
  const int bN    = blockIdx.y * 128;

  const int ldRow = tid >> 1;         // 0..127
  const int ldSeg = (tid & 1) * 16;   // 0 or 16 (elements)

  const AT* aRow = A + (size_t)(bM + ldRow) * Dm + ldSeg;
  const float* wRow = W + (size_t)(bN + ldRow) * Dm + ldSeg;

  v8f acc[2][4];
#pragma unroll
  for (int i = 0; i < 2; ++i)
#pragma unroll
    for (int j = 0; j < 4; ++j)
      acc[i][j] = (v8f){0.f, 0.f, 0.f, 0.f, 0.f, 0.f, 0.f, 0.f};

  for (int k0 = 0; k0 < Dm; k0 += 32) {
    __syncthreads();
    // ---- stage A tile (convert fp32->bf16 if needed) ----
    {
      uint4* dst = (uint4*)&sA[ldRow * LST + ldSeg];
      if constexpr (sizeof(AT) == 4) {
        const float4* p = (const float4*)(aRow + k0);
        float4 f0 = p[0], f1 = p[1], f2 = p[2], f3 = p[3];
        dst[0] = make_uint4(pack2f(f0.x, f0.y), pack2f(f0.z, f0.w),
                            pack2f(f1.x, f1.y), pack2f(f1.z, f1.w));
        dst[1] = make_uint4(pack2f(f2.x, f2.y), pack2f(f2.z, f2.w),
                            pack2f(f3.x, f3.y), pack2f(f3.z, f3.w));
      } else {
        const uint4* p = (const uint4*)(aRow + k0);
        dst[0] = p[0];
        dst[1] = p[1];
      }
    }
    // ---- stage W tile (always fp32 -> bf16) ----
    {
      const float4* p = (const float4*)(wRow + k0);
      float4 f0 = p[0], f1 = p[1], f2 = p[2], f3 = p[3];
      uint4* dst = (uint4*)&sW[ldRow * LST + ldSeg];
      dst[0] = make_uint4(pack2f(f0.x, f0.y), pack2f(f0.z, f0.w),
                          pack2f(f1.x, f1.y), pack2f(f1.z, f1.w));
      dst[1] = make_uint4(pack2f(f2.x, f2.y), pack2f(f2.z, f2.w),
                          pack2f(f3.x, f3.y), pack2f(f3.z, f3.w));
    }
    // ---- prefetch next K-tile rows into cache while we compute ----
    if (k0 + 32 < Dm) {
      __builtin_prefetch(aRow + k0 + 32, 0, 1);
      __builtin_prefetch(wRow + k0 + 32, 0, 1);
    }
    __syncthreads();

    // ---- fragments + 8 WMMAs ----
    FragBF afr[2], bfr[4];
#pragma unroll
    for (int mf = 0; mf < 2; ++mf) {
      const int row = waveM + mf * 16 + lrow;
#pragma unroll
      for (int i = 0; i < 8; ++i)
        afr[mf].u[i] = *(const unsigned*)&sA[row * LST + kpat(i, hi)];
    }
#pragma unroll
    for (int nf = 0; nf < 4; ++nf) {
      const int row = waveN + nf * 16 + lrow;
#pragma unroll
      for (int i = 0; i < 8; ++i)
        bfr[nf].u[i] = *(const unsigned*)&sW[row * LST + kpat(i, hi)];
    }
#pragma unroll
    for (int mf = 0; mf < 2; ++mf)
#pragma unroll
      for (int nf = 0; nf < 4; ++nf)
        acc[mf][nf] = wmma_bf16(afr[mf], bfr[nf], acc[mf][nf]);
  }

  // ---- epilogue: bias + store (C layout: VGPR r -> row r + hi*8, lane -> col)
#pragma unroll
  for (int mf = 0; mf < 2; ++mf) {
    const int m0 = bM + waveM + mf * 16 + hi * 8;
#pragma unroll
    for (int nf = 0; nf < 4; ++nf) {
      const int n = bN + waveN + nf * 16 + lrow;
      const float bv = bias[n];
#pragma unroll
      for (int r = 0; r < 8; ++r) {
        const float val = acc[mf][nf][r] + bv;
        if constexpr (OUT_MODE == 0) {
          ((float*)Cout)[(size_t)(m0 + r) * Dm + n] = val;
        } else if constexpr (OUT_MODE == 1) {
          ((__bf16*)Cout)[(size_t)(m0 + r) * Dm + n] = (__bf16)val;
        } else {
          const int m = m0 + r;          // global token row in [0, 16384)
          const int bb = m >> 11;        // batch
          const int t  = m & 2047;       // token within batch
          ((__bf16*)Cout)[((size_t)bb * 1024 + n) * 2048 + t] = (__bf16)val;
        }
      }
    }
  }
}

// ---------------------------------------------------------------------------
// Flash attention: one block per (b, h, 64-row q tile), 128 threads (4 waves),
// each wave owns 16 q rows. K-tiles of 64, online softmax, WMMA for QK^T & PV.
// K and V^T tiles staged with gfx1250 GLOBAL_LOAD_ASYNC_TO_LDS_B128.
// Faithful to reference: masked scores = -1e-9 (still inside softmax).
// ---------------------------------------------------------------------------
__global__ __launch_bounds__(128) void attn_kernel(
    const __bf16* __restrict__ Qb, const __bf16* __restrict__ Kb,
    const __bf16* __restrict__ Vt, const int* __restrict__ dec_len,
    const int* __restrict__ enc_len, __bf16* __restrict__ Ob) {
  constexpr int TFUT = 512, TLOOK = 2048, HD = 64;
  constexpr int LST = 72;
  __shared__ __bf16 sK[64 * LST];   // sK[kpos][hd]
  __shared__ __bf16 sVt[64 * LST];  // sVt[hd][kpos] (V^T, staged directly)
  __shared__ __bf16 sP[64 * LST];   // sP[q_local][kpos], per-wave 16-row slices

  const int tid  = threadIdx.x;
  const int lane = tid & 31;
  const int wid  = tid >> 5;
  const int hi   = (lane >> 4) & 1;
  const int lrow = lane & 15;

  const int bid = blockIdx.x;
  const int qt  = bid & 7;          // 8 q-tiles of 64
  const int h   = (bid >> 3) & 15;  // 16 heads
  const int b   = bid >> 7;         // 8 batches

  const int dl = dec_len[b];
  const int el = enc_len[b];
  const int qloc = qt * 64 + wid * 16 + hi * 8;  // + r gives q index in T_FUT

  // staging assignment: thread -> (row r0, 32-element segment sg)
  const int r0 = tid >> 1;
  const int sg = (tid & 1) * 32;
  const __bf16* gK0 = Kb + (size_t)(b * TLOOK + r0) * Dm + h * HD + sg;
  const __bf16* gV0 = Vt + ((size_t)(b * 16 + h) * HD + r0) * TLOOK + sg;
  const unsigned dK = (unsigned)(uintptr_t)&sK[r0 * LST + sg];
  const unsigned dV = (unsigned)(uintptr_t)&sVt[r0 * LST + sg];

  // ---- load Q fragments once (A layout), K dim = 64 -> 2 chunks of 32 ----
  FragBF qa[2];
  {
    const int qrow = qt * 64 + wid * 16 + lrow;
    const __bf16* qp = Qb + (size_t)(b * TFUT + qrow) * Dm + h * HD;
#pragma unroll
    for (int c = 0; c < 2; ++c)
#pragma unroll
      for (int i = 0; i < 8; ++i)
        qa[c].u[i] = *(const unsigned*)(qp + c * 32 + kpat(i, hi));
  }

  v8f oacc[4];
#pragma unroll
  for (int nf = 0; nf < 4; ++nf)
    oacc[nf] = (v8f){0.f, 0.f, 0.f, 0.f, 0.f, 0.f, 0.f, 0.f};
  float m_r[8], l_r[8];
#pragma unroll
  for (int r = 0; r < 8; ++r) { m_r[r] = -3.0e38f; l_r[r] = 0.f; }

  for (int kt = 0; kt < TLOOK; kt += 64) {
    __syncthreads();  // previous tile fully consumed by all waves
    // ---- async stage K tile: sK[kpos][hd] (64 x 64 bf16) ----
#pragma unroll
    for (int j = 0; j < 4; ++j)
      async_ld_b128(dK + j * 16, gK0 + (size_t)kt * Dm + j * 8);
    // ---- async stage V^T tile: sVt[hd][kpos] (64 x 64 bf16) ----
#pragma unroll
    for (int j = 0; j < 4; ++j)
      async_ld_b128(dV + j * 16, gV0 + kt + j * 8);
    wait_asynccnt0();
    __syncthreads();

    // ---- scores S[16 q][64 k] = Q K^T : 4 nfrags x 2 k-chunks of WMMA ----
    v8f S[4];
#pragma unroll
    for (int nf = 0; nf < 4; ++nf) {
      S[nf] = (v8f){0.f, 0.f, 0.f, 0.f, 0.f, 0.f, 0.f, 0.f};
#pragma unroll
      for (int c = 0; c < 2; ++c) {
        FragBF kf;
#pragma unroll
        for (int i = 0; i < 8; ++i)
          kf.u[i] = *(const unsigned*)&sK[(nf * 16 + lrow) * LST + c * 32 + kpat(i, hi)];
        S[nf] = wmma_bf16(qa[c], kf, S[nf]);
      }
    }

    // ---- mask + scale (reference: where(mask, s/8, -1e-9)) ----
#pragma unroll
    for (int nf = 0; nf < 4; ++nf) {
      const int kp = kt + nf * 16 + lrow;
      const bool kv = kp < el;
#pragma unroll
      for (int r = 0; r < 8; ++r) {
        const bool qv = (qloc + r) < dl;
        const float s = S[nf][r];
        S[nf][r] = (qv && kv) ? s * 0.125f : -1e-9f;
      }
    }

    // ---- online softmax ----
    float mn[8], corr[8];
#pragma unroll
    for (int r = 0; r < 8; ++r) {
      float mx = fmaxf(fmaxf(S[0][r], S[1][r]), fmaxf(S[2][r], S[3][r]));
      mx = fmaxf(mx, __shfl_xor(mx, 1, 32));
      mx = fmaxf(mx, __shfl_xor(mx, 2, 32));
      mx = fmaxf(mx, __shfl_xor(mx, 4, 32));
      mx = fmaxf(mx, __shfl_xor(mx, 8, 32));
      mn[r] = fmaxf(m_r[r], mx);
      corr[r] = __expf(m_r[r] - mn[r]);
      m_r[r] = mn[r];
    }
#pragma unroll
    for (int nf = 0; nf < 4; ++nf)
#pragma unroll
      for (int r = 0; r < 8; ++r)
        S[nf][r] = __expf(S[nf][r] - mn[r]);
#pragma unroll
    for (int r = 0; r < 8; ++r) {
      float sm = S[0][r] + S[1][r] + S[2][r] + S[3][r];
      sm += __shfl_xor(sm, 1, 32);
      sm += __shfl_xor(sm, 2, 32);
      sm += __shfl_xor(sm, 4, 32);
      sm += __shfl_xor(sm, 8, 32);
      l_r[r] = l_r[r] * corr[r] + sm;
    }
#pragma unroll
    for (int nf = 0; nf < 4; ++nf)
#pragma unroll
      for (int r = 0; r < 8; ++r)
        oacc[nf][r] *= corr[r];

    // ---- write P (bf16) into per-wave LDS slice, reload in A layout ----
#pragma unroll
    for (int nf = 0; nf < 4; ++nf)
#pragma unroll
      for (int r = 0; r < 8; ++r)
        sP[(wid * 16 + r + hi * 8) * LST + nf * 16 + lrow] = (__bf16)S[nf][r];
    asm volatile("s_wait_dscnt 0" ::: "memory");  // same-wave LDS RAW

    FragBF pf[2];
#pragma unroll
    for (int c = 0; c < 2; ++c)
#pragma unroll
      for (int i = 0; i < 8; ++i)
        pf[c].u[i] = *(const unsigned*)&sP[(wid * 16 + lrow) * LST + c * 32 + kpat(i, hi)];

    // ---- O += P V : 4 hd-frags x 2 kpos-chunks ----
#pragma unroll
    for (int nf = 0; nf < 4; ++nf)
#pragma unroll
      for (int c = 0; c < 2; ++c) {
        FragBF vf;
#pragma unroll
        for (int i = 0; i < 8; ++i)
          vf.u[i] = *(const unsigned*)&sVt[(nf * 16 + lrow) * LST + c * 32 + kpat(i, hi)];
        oacc[nf] = wmma_bf16(pf[c], vf, oacc[nf]);
      }
  }

  // ---- normalize + store attended (bf16) ----
#pragma unroll
  for (int r = 0; r < 8; ++r) {
    const float inv = 1.f / l_r[r];
    const size_t row = (size_t)(b * TFUT + qloc + r) * Dm + h * HD;
#pragma unroll
    for (int nf = 0; nf < 4; ++nf)
      Ob[row + nf * 16 + lrow] = (__bf16)(oacc[nf][r] * inv);
  }
}

// ---------------------------------------------------------------------------
extern "C" void kernel_launch(void* const* d_in, const int* in_sizes, int n_in,
                              void* d_out, int out_size, void* d_ws, size_t ws_size,
                              hipStream_t stream) {
  (void)in_sizes; (void)n_in; (void)out_size; (void)ws_size;
  const float* enc  = (const float*)d_in[0];   // [8,2048,1024]
  const float* fq   = (const float*)d_in[1];   // [8,512,1024]
  const int*   dlen = (const int*)d_in[2];
  const int*   elen = (const int*)d_in[3];
  const float* wq_w = (const float*)d_in[4];
  const float* wq_b = (const float*)d_in[5];
  const float* wk_w = (const float*)d_in[6];
  const float* wk_b = (const float*)d_in[7];
  const float* wv_w = (const float*)d_in[8];
  const float* wv_b = (const float*)d_in[9];
  const float* wo_w = (const float*)d_in[10];
  const float* wo_b = (const float*)d_in[11];
  float* out = (float*)d_out;

  char* ws = (char*)d_ws;
  __bf16* Qb = (__bf16*)(ws);                              //  8 MiB: 4096x1024
  __bf16* Kb = (__bf16*)(ws + (size_t)8  * 1024 * 1024);   // 32 MiB: 16384x1024
  __bf16* Vt = (__bf16*)(ws + (size_t)40 * 1024 * 1024);   // 32 MiB (per-head V^T)
  __bf16* Ob = (__bf16*)(ws + (size_t)72 * 1024 * 1024);   //  8 MiB

  // Q/K projections (fp32 in -> bf16 out row-major); V -> bf16 per-head V^T
  gemm_bias_kernel<float, 1><<<dim3(4096 / 128, 8), 256, 0, stream>>>(fq, wq_w, wq_b, Qb);
  gemm_bias_kernel<float, 1><<<dim3(16384 / 128, 8), 256, 0, stream>>>(enc, wk_w, wk_b, Kb);
  gemm_bias_kernel<float, 2><<<dim3(16384 / 128, 8), 256, 0, stream>>>(enc, wv_w, wv_b, Vt);

  // attention: 8 batches * 16 heads * 8 q-tiles = 1024 blocks
  attn_kernel<<<1024, 128, 0, stream>>>(Qb, Kb, Vt, dlen, elen, Ob);

  // output projection (bf16 in -> fp32 out)
  gemm_bias_kernel<__bf16, 0><<<dim3(4096 / 128, 8), 256, 0, stream>>>(Ob, wo_w, wo_b, out);
}
